// BudgetAttentionTwo_52793738002651
// MI455X (gfx1250) — matrix operand order repaired and checked by reading, not compiled, not run on hardware
//
#include <hip/hip_runtime.h>
#include <hip/hip_bf16.h>

typedef __attribute__((ext_vector_type(16))) _Float16 v16h;
typedef __attribute__((ext_vector_type(8)))  _Float16 v8h;
typedef __attribute__((ext_vector_type(8)))  float    v8f;

#define B_ 4
#define N_ 2048
#define E_ 512
#define H_ 8
#define D_ 64

// sqrt(log2(e)): folded into K so scores arrive pre-scaled for exp2.
#define SQRT_LOG2E 1.2011224087864498f

// A-matrix fragment (16x32, 16-bit): lane holds row (lane&15); half k-pattern
// is {8*hi+0..7, 16+8*hi+0..7}. p = row base + (hi<<3).
__device__ __forceinline__ v16h load_frag(const _Float16* p) {
  v8h lo = *(const v8h*)(p);
  v8h hi = *(const v8h*)(p + 16);
  v16h a;
#pragma unroll
  for (int i = 0; i < 8; ++i) { a[i] = lo[i]; a[i + 8] = hi[i]; }
  return a;
}

// B-matrix fragment (32x16, 16-bit): lane holds column (lane&15); half
// k-pattern is {8*hi+0..7, 16+8*hi+0..7}. p = col base + (hi<<4).
__device__ __forceinline__ v16h load_bfrag(const _Float16* p) {
  p -= ((threadIdx.x & 31) >> 4) << 3;
  v8h lo = *(const v8h*)(p);
  v8h hi = *(const v8h*)(p + 16);
  v16h b;
#pragma unroll
  for (int i = 0; i < 8; ++i) { b[i] = lo[i]; b[i + 8] = hi[i]; }
  return b;
}

__device__ __forceinline__ v8f wmma16(v16h a, v16h b, v8f c) {
  return __builtin_amdgcn_wmma_f32_16x16x32_f16(false, a, false, b, (short)0, c,
                                                false, false);
}

__global__ void cvt_f32_f16_kernel(const float* __restrict__ src,
                                   _Float16* __restrict__ dst, int n) {
  int i = blockIdx.x * blockDim.x + threadIdx.x;
  int stride = gridDim.x * blockDim.x;
  for (; i < n; i += stride) dst[i] = (_Float16)src[i];
}

// One wave computes one 16x16 tile of BOTH K = x@Wk^T + bk and V = x@Wv^T + bv.
// K stored f16 as [b,h,n,d] (d contiguous), pre-scaled by sqrt(log2e);
// V stored f16 transposed [b,h,d,n].
__global__ void __launch_bounds__(128)
proj_kernel(const _Float16* __restrict__ xh, const _Float16* __restrict__ wkh,
            const _Float16* __restrict__ wvh, const float* __restrict__ bk,
            const float* __restrict__ bv, _Float16* __restrict__ Kh,
            _Float16* __restrict__ Vth) {
  const int lane = threadIdx.x & 31;
  const int wid  = threadIdx.x >> 5;
  const int w    = blockIdx.x * 4 + wid;       // 0..16383 tiles
  const int nt   = w & 31;                     // 32 tiles over E
  const int mt   = w >> 5;                     // 512 tiles over B*N
  const int m15  = lane & 15;
  const int hi   = lane >> 4;
  const int kh8  = hi << 3;                    // A-fragment base offset
  const int kh16 = hi << 4;                    // B-fragment base offset

  const _Float16* ap  = xh  + (size_t)((mt << 4) + m15) * E_ + kh8;
  const _Float16* bkp = wkh + (size_t)((nt << 4) + m15) * E_ + kh16;
  const _Float16* bvp = wvh + (size_t)((nt << 4) + m15) * E_ + kh16;

  v8f accK = {};
  v8f accV = {};
#pragma unroll 4
  for (int kk = 0; kk < E_; kk += 32) {
    v16h a  = load_frag(ap + kk);
    v16h b0 = load_bfrag(bkp + kk);
    v16h b1 = load_bfrag(bvp + kk);
    accK = wmma16(a, b0, accK);
    accV = wmma16(a, b1, accV);
  }

  const int ncol = (nt << 4) + m15;            // output feature e
  const float biasK = bk[ncol];
  const float biasV = bv[ncol];
  const int h = ncol >> 6;                     // head
  const int d = ncol & 63;                     // dim within head
#pragma unroll
  for (int r = 0; r < 8; ++r) {
    int row  = (mt << 4) + (hi << 3) + r;      // flat token index in [0, B*N)
    int bidx = row >> 11;                      // N_ = 2048
    int ntok = row & (N_ - 1);
    size_t kb = ((size_t)(bidx * H_ + h) * N_ + ntok) * D_ + d;
    size_t vb = ((size_t)(bidx * H_ + h) * D_ + d) * N_ + ntok;
    Kh[kb]  = (_Float16)((accK[r] + biasK) * SQRT_LOG2E);
    Vth[vb] = (_Float16)(accV[r] + biasV);
  }
}

// Flash attention: one wave owns 16 query rows of one (b,h); 64-col key tiles.
// S = K K^T (WMMA f16, pre-scaled by log2e), online softmax in fp32 with
// exp2, row sums via P @ ones WMMA, O += P @ V (WMMA f16).
__global__ void __launch_bounds__(128)
attn_kernel(const _Float16* __restrict__ Kh, const _Float16* __restrict__ Vth,
            float* __restrict__ out) {
  __shared__ __align__(16) _Float16 plds[4 * 16 * 64];  // per-wave 16x64 P tile
  const int lane = threadIdx.x & 31;
  const int wid  = threadIdx.x >> 5;
  const int w    = blockIdx.x * 4 + wid;       // 0..4095 row tiles
  const int bh   = w >> 7;                     // (b*H + h) in [0,32)
  const int i0   = (w & 127) << 4;             // query row block
  const int bidx = bh >> 3;
  const int h    = bh & 7;
  const int m15  = lane & 15;
  const int hi   = lane >> 4;
  const int kh8  = hi << 3;
  const int kh16 = hi << 4;

  const _Float16* Kb  = Kh  + (size_t)bh * N_ * D_;
  const _Float16* Vtb = Vth + (size_t)bh * D_ * N_;
  _Float16* pl = plds + wid * (16 * 64);

  // Q-fragments (Q == K rows), K-dim = D = 64 -> two k=32 A-fragments.
  const _Float16* arow = Kb + (size_t)(i0 + m15) * D_ + kh8;
  v16h a0 = load_frag(arow);
  v16h a1 = load_frag(arow + 32);

  v16h ones;
#pragma unroll
  for (int i = 0; i < 16; ++i) ones[i] = (_Float16)1.0f;

  float rmax[8];
#pragma unroll
  for (int r = 0; r < 8; ++r) rmax[r] = -3.0e30f;
  v8f o0 = {}, o1 = {}, o2 = {}, o3 = {}, osum = {};

  for (int j0 = 0; j0 < N_; j0 += 64) {
    // ---- scores: S[16, j0..j0+64) in 4 accumulators ----
    v8f s[4];
    const _Float16* cb = Kb + (size_t)(j0 + m15) * D_ + kh16;
#pragma unroll
    for (int t = 0; t < 4; ++t) {
      const _Float16* c = cb + (size_t)(t * 16) * D_;
      v16h bk0 = load_bfrag(c);        // key cols, d = 0..31
      v16h bk1 = load_bfrag(c + 32);   // key cols, d = 32..63
      v8f acc = {};
      acc = wmma16(a0, bk0, acc);
      acc = wmma16(a1, bk1, acc);
      s[t] = acc;
    }

    // ---- online softmax; rows split lanes 0-15 / 16-31; exp2 domain ----
    float corr[8];
#pragma unroll
    for (int r = 0; r < 8; ++r) {
      float t0 = fmaxf(fmaxf(s[0][r], s[1][r]), fmaxf(s[2][r], s[3][r]));
      t0 = fmaxf(t0, __shfl_xor(t0, 1, 32));
      t0 = fmaxf(t0, __shfl_xor(t0, 2, 32));
      t0 = fmaxf(t0, __shfl_xor(t0, 4, 32));
      t0 = fmaxf(t0, __shfl_xor(t0, 8, 32));
      float nm = fmaxf(rmax[r], t0);
      corr[r] = __builtin_amdgcn_exp2f(rmax[r] - nm);
      rmax[r] = nm;
      int rowoff = ((hi << 3) + r) * 64 + m15;
#pragma unroll
      for (int t = 0; t < 4; ++t) {
        float p = __builtin_amdgcn_exp2f(s[t][r] - nm);
        pl[rowoff + t * 16] = (_Float16)p;   // P tile -> LDS (f16)
      }
    }
#pragma unroll
    for (int r = 0; r < 8; ++r) {
      o0[r] *= corr[r]; o1[r] *= corr[r]; o2[r] *= corr[r]; o3[r] *= corr[r];
      osum[r] *= corr[r];
    }

    // ---- P: accumulator layout -> A-fragment layout via per-wave LDS ----
    asm volatile("" ::: "memory");  // LDS pipe is in-order within a wave
    v16h pf0 = load_frag(pl + m15 * 64 + kh8);
    v16h pf1 = load_frag(pl + m15 * 64 + 32 + kh8);
    asm volatile("" ::: "memory");

    // ---- running row sums on the tensor core: osum += P @ ones ----
    osum = wmma16(pf0, ones, osum);
    osum = wmma16(pf1, ones, osum);

    // ---- O += P @ V ; V^T rows are contiguous in j ----
    const _Float16* vb = Vtb + j0 + kh16;
    {
      v16h v0a = load_bfrag(vb + (size_t)(m15)      * N_);
      v16h v0b = load_bfrag(vb + (size_t)(m15)      * N_ + 32);
      o0 = wmma16(pf0, v0a, o0);
      o0 = wmma16(pf1, v0b, o0);
      v16h v1a = load_bfrag(vb + (size_t)(m15 + 16) * N_);
      v16h v1b = load_bfrag(vb + (size_t)(m15 + 16) * N_ + 32);
      o1 = wmma16(pf0, v1a, o1);
      o1 = wmma16(pf1, v1b, o1);
      v16h v2a = load_bfrag(vb + (size_t)(m15 + 32) * N_);
      v16h v2b = load_bfrag(vb + (size_t)(m15 + 32) * N_ + 32);
      o2 = wmma16(pf0, v2a, o2);
      o2 = wmma16(pf1, v2b, o2);
      v16h v3a = load_bfrag(vb + (size_t)(m15 + 48) * N_);
      v16h v3b = load_bfrag(vb + (size_t)(m15 + 48) * N_ + 32);
      o3 = wmma16(pf0, v3a, o3);
      o3 = wmma16(pf1, v3b, o3);
    }
  }

  // Final: divide by row sum, apply reference's post-softmax 1/sqrt(E).
  const float inv_sqrtE = 0.044194173824159216f;  // 1/sqrt(512)
#pragma unroll
  for (int r = 0; r < 8; ++r) {
    float sc = inv_sqrtE / osum[r];   // every column of osum holds the row sum
    int row = i0 + (hi << 3) + r;
    float* op = out + ((size_t)bidx * N_ + row) * E_ + h * 64 + m15;
    op[0]  = o0[r] * sc;
    op[16] = o1[r] * sc;
    op[32] = o2[r] * sc;
    op[48] = o3[r] * sc;
  }
}

extern "C" void kernel_launch(void* const* d_in, const int* in_sizes, int n_in,
                              void* d_out, int out_size, void* d_ws, size_t ws_size,
                              hipStream_t stream) {
  (void)in_sizes; (void)n_in; (void)out_size; (void)ws_size;
  const float* x  = (const float*)d_in[0];
  const float* Wk = (const float*)d_in[1];
  const float* bk = (const float*)d_in[2];
  const float* Wv = (const float*)d_in[3];
  const float* bv = (const float*)d_in[4];
  float* out = (float*)d_out;

  // Workspace layout (halves): x_h | Wk_h | Wv_h | K_h | Vt_h  (~25 MB)
  _Float16* ws  = (_Float16*)d_ws;
  _Float16* xh  = ws;                       // 8192*512       = 4194304
  _Float16* wkh = xh  + 4194304;            // 512*512        =  262144
  _Float16* wvh = wkh + 262144;
  _Float16* Kh  = wvh + 262144;             // B*H*N*D        = 4194304
  _Float16* Vth = Kh  + 4194304;            // B*H*D*N        = 4194304

  cvt_f32_f16_kernel<<<2048, 256, 0, stream>>>(x, xh, 4194304);
  cvt_f32_f16_kernel<<<256, 256, 0, stream>>>(Wk, wkh, 262144);
  cvt_f32_f16_kernel<<<256, 256, 0, stream>>>(Wv, wvh, 262144);

  // 512 M-tiles x 32 N-tiles, one wave per tile, 4 waves/block.
  proj_kernel<<<4096, 128, 0, stream>>>(xh, wkh, wvh, bk, bv, Kh, Vth);

  // 32 (b,h) x 128 row-tiles, one wave per 16 rows, 4 waves/block.
  attn_kernel<<<1024, 128, 0, stream>>>(Kh, Vth, out);
}
